// MemoryBlock_15144054686530
// MI455X (gfx1250) — matrix operand order repaired and checked by reading, not compiled, not run on hardware
//
#include <hip/hip_runtime.h>
#include <hip/hip_bf16.h>
#include <math.h>

// ---------------------------------------------------------------------------
// MemoryBlock (TTT-style chunked memory) for MI455X / gfx1250, wave32 + WMMA.
// All matmul operands are pre-converted bf16, stored K-contiguous for their
// consumer so every fragment load is two global_load_b128. fp32 masters are
// kept only for the W1/W2 carries (L2-resident) and scalar epilogues.
// Each wave processes two (scan) or four (QKV proj) 16x16 output tiles sharing one A fragment.
// ---------------------------------------------------------------------------

constexpr int D    = 1024;
constexpr int S    = 4096;
constexpr int BB   = 4;
constexpr int CH   = 16;
constexpr int NCH  = S / CH;   // 256 scan steps

typedef __attribute__((ext_vector_type(16))) __bf16 bf16x16;
typedef __attribute__((ext_vector_type(8)))  __bf16 bf16x8;
typedef __attribute__((ext_vector_type(8)))  float  f32x8;

__device__ inline int laneid() { return (int)(threadIdx.x & 31); }

__device__ inline f32x8 zero8() {
  f32x8 z;
#pragma unroll
  for (int i = 0; i < 8; ++i) z[i] = 0.0f;
  return z;
}

__device__ inline f32x8 wmma_bf16(bf16x16 a, bf16x16 b, f32x8 c) {
  return __builtin_amdgcn_wmma_f32_16x16x32_bf16(false, a, false, b,
                                                 (short)0, c, false, false);
}

__device__ inline bf16x16 frag_cat(bf16x8 lo, bf16x8 hi) {
  bf16x16 r;
#pragma unroll
  for (int i = 0; i < 8; ++i) { r[i] = lo[i]; r[8 + i] = hi[i]; }
  return r;
}

// ---- fragment loaders (ISA 7.12.2 layouts, wave32) ------------------------
// A (16x32): lane l<16 row M=l, K {kh..kh+7, 16+kh..16+kh+7}, kh = 8*(l>=16).
// Source row-major bf16 [16][ld], K contiguous -> two 16B loads.
__device__ inline bf16x16 load_a(const __bf16* base, int ld, int kb) {
  int l = laneid();
  const __bf16* p = base + (size_t)(l & 15) * ld + kb + ((l >> 4) * 8);
  return frag_cat(*(const bf16x8*)p, *(const bf16x8*)(p + 16));
}

// B (32x16): lane l holds col N=l&15, K {kh..kh+7, 16+kh..16+kh+7}, kh = 8*(l>=16).
// Source n-major bf16 (element B[k][n] = base[n*ld + k]) -> two 16B loads.
__device__ inline bf16x16 load_b(const __bf16* base, int ld, int kb) {
  int l = laneid();
  const __bf16* p = base + (size_t)(l & 15) * ld + kb + ((l >> 4) * 8);
  return frag_cat(*(const bf16x8*)p, *(const bf16x8*)(p + 16));
}

// A fragment, K=16 (rank-16 update): elements 0..7 are K=kh..kh+7 (<16, valid
// for both lane halves), elements 8..15 are K>=16 -> zero. Source [16][16].
__device__ inline bf16x16 load_a_k16(const __bf16* base) {
  int l = laneid();
  const __bf16* p = base + (size_t)(l & 15) * 16 + ((l >> 4) * 8);
  bf16x8 lo = *(const bf16x8*)p;
  bf16x16 r;
#pragma unroll
  for (int i = 0; i < 8; ++i) { r[i] = lo[i]; r[8 + i] = (__bf16)0.0f; }
  return r;
}

// B fragment, K=16: elements 0..7 are K=kh..kh+7, elements 8..15 (K>=16) zero. Source [16][16] n-major.
__device__ inline bf16x16 load_b_k16(const __bf16* base) {
  int l = laneid();
  const __bf16* p = base + (size_t)(l & 15) * 16 + ((l >> 4) * 8);
  bf16x8 lo = *(const bf16x8*)p;
  bf16x16 r;
#pragma unroll
  for (int i = 0; i < 8; ++i) { r[i] = lo[i]; r[8 + i] = (__bf16)0.0f; }
  return r;
}

// pack 8 accum lanes -> 16B store (used for transposed bf16 epilogues)
__device__ inline bf16x8 pack8(const f32x8& a) {
  bf16x8 p;
#pragma unroll
  for (int i = 0; i < 8; ++i) p[i] = (__bf16)a[i];
  return p;
}

// ---- activation helpers ---------------------------------------------------
__device__ inline float sigmoid_f(float x) { return 1.0f / (1.0f + expf(-x)); }

__device__ inline float gelu_f(float x) {   // tanh approximation (jax default)
  const float kA = 0.7978845608028654f;
  float u = kA * (x + 0.044715f * x * x * x);
  return 0.5f * x * (1.0f + tanhf(u));
}

__device__ inline float dgelu_f(float x) {
  const float kA = 0.7978845608028654f;
  float x2 = x * x;
  float u = kA * (x + 0.044715f * x * x2);
  float t = tanhf(u);
  float s = 1.0f - t * t;
  return 0.5f * (1.0f + t) + 0.5f * x * s * kA * (1.0f + 0.134145f * x2);
}

// ===========================================================================
// Pre-pass A: x -> bf16 (vectorized). grid = BSD/2048.
// ===========================================================================
__global__ void cvt_x_kernel(const float* __restrict__ x, __bf16* __restrict__ xb) {
  size_t i = ((size_t)blockIdx.x * 256 + threadIdx.x) * 8;
  float4 a = *(const float4*)(x + i);
  float4 b = *(const float4*)(x + i + 4);
  bf16x8 o;
  o[0] = (__bf16)a.x; o[1] = (__bf16)a.y; o[2] = (__bf16)a.z; o[3] = (__bf16)a.w;
  o[4] = (__bf16)b.x; o[5] = (__bf16)b.y; o[6] = (__bf16)b.z; o[7] = (__bf16)b.w;
  *(bf16x8*)(xb + i) = o;
}

// ===========================================================================
// Pre-pass B: Wq/Wk/Wv -> transposed bf16 Wt[z][n*D+k]. grid = (D*D/256, 3).
// ===========================================================================
__global__ void wt_kernel(const float* __restrict__ Wq,
                          const float* __restrict__ Wk,
                          const float* __restrict__ Wv,
                          __bf16* __restrict__ Wt) {
  size_t i = (size_t)blockIdx.x * 256 + threadIdx.x;   // over D*D, i = n*D+k
  int n = (int)(i / D), k = (int)(i % D);
  const float* W = (blockIdx.y == 0) ? Wq : (blockIdx.y == 1) ? Wk : Wv;
  Wt[(size_t)blockIdx.y * D * D + i] = (__bf16)W[(size_t)k * D + n];
}

// ===========================================================================
// Pre-pass C: broadcast W1/W2 into fp32 carries + bf16 shadows.
//   w1t[e][d], w2rm[e][f], w2t[f][e].  grid = (D*D/256, B, 2).
// ===========================================================================
__global__ void init_w_kernel(const float* __restrict__ W1,
                              const float* __restrict__ W2,
                              float* __restrict__ w1b, float* __restrict__ w2b,
                              __bf16* __restrict__ w1t,
                              __bf16* __restrict__ w2rm,
                              __bf16* __restrict__ w2t) {
  size_t i = (size_t)blockIdx.x * 256 + threadIdx.x;
  int b = (int)blockIdx.y;
  int r = (int)(i / D), cc = (int)(i % D);
  size_t off = (size_t)b * D * D;
  if (blockIdx.z == 0) {
    float v = W1[i];
    w1b[off + i] = v;
    w1t[off + (size_t)cc * D + r] = (__bf16)v;
  } else {
    float v = W2[i];
    w2b[off + i] = v;
    w2rm[off + i] = (__bf16)v;
    w2t[off + (size_t)cc * D + r] = (__bf16)v;
  }
}

// ===========================================================================
// QKV projection: grid = (BS/128, D/64, 3); 8 waves, wave = 16 rows x 64 cols.
// z==1 (k) additionally emits per-chunk transposed tiles kt[chunk][d][c].
// ===========================================================================
__global__ void proj_kernel(const __bf16* __restrict__ xb,
                            const __bf16* __restrict__ Wt,
                            __bf16* __restrict__ qb,
                            __bf16* __restrict__ kb,
                            __bf16* __restrict__ vb,
                            __bf16* __restrict__ kt) {
  const __bf16* W = Wt + (size_t)blockIdx.z * D * D;
  __bf16* dst = (blockIdx.z == 0) ? qb : (blockIdx.z == 1) ? kb : vb;

  int wave = (int)(threadIdx.x >> 5);
  int mt = (int)blockIdx.x * 8 + wave;          // 16-row tile over B*S (=chunk)
  int nb = (int)blockIdx.y * 64;
  const __bf16* Abase = xb + (size_t)mt * 16 * D;

  f32x8 acc[4];
#pragma unroll
  for (int t = 0; t < 4; ++t) acc[t] = zero8();

  for (int kk = 0; kk < D; kk += 32) {
    if (kk + 64 < D) __builtin_prefetch(Abase + kk + 64, 0, 3);
    bf16x16 af = load_a(Abase, D, kk);
#pragma unroll
    for (int t = 0; t < 4; ++t) {
      bf16x16 bf = load_b(W + (size_t)(nb + t * 16) * D, D, kk);
      acc[t] = wmma_bf16(af, bf, acc[t]);
    }
  }
  int l = laneid(), n0 = l & 15, mo = (l >> 4) * 8;
#pragma unroll
  for (int t = 0; t < 4; ++t) {
    int col = nb + t * 16 + n0;
#pragma unroll
    for (int r = 0; r < 8; ++r)
      dst[(size_t)(mt * 16 + mo + r) * D + col] = (__bf16)acc[t][r];
    if (blockIdx.z == 1)
      *(bf16x8*)(kt + ((size_t)mt * D + col) * 16 + mo) = pack8(acc[t]);
  }
}

// ===========================================================================
// per-token lr + per-chunk decay. grid = B*NCH, 256 threads.
// ===========================================================================
__global__ void lrdecay_kernel(const float* __restrict__ x,
                               const float* __restrict__ wlr,
                               const float* __restrict__ wdec,
                               float* __restrict__ lr,
                               float* __restrict__ decc) {
  int chunk = (int)blockIdx.x;
  int b = chunk / NCH, n = chunk % NCH;
  int wave = (int)(threadIdx.x >> 5), l = laneid();
  __shared__ float sdec[CH];

  for (int j = 0; j < 2; ++j) {
    int cloc = wave * 2 + j;
    const float* xr = x + ((size_t)b * S + (size_t)n * CH + cloc) * D;
    float s1 = 0.0f, s2 = 0.0f;
    for (int i = l; i < D; i += 32) {
      float xv = xr[i];
      s1 += xv * wlr[i];
      s2 += xv * wdec[i];
    }
#pragma unroll
    for (int off = 16; off >= 1; off >>= 1) {
      s1 += __shfl_down(s1, off, 32);
      s2 += __shfl_down(s2, off, 32);
    }
    if (l == 0) {
      lr[(size_t)b * S + n * CH + cloc] = 0.1f * sigmoid_f(s1);
      sdec[cloc] = sigmoid_f(s2);
    }
  }
  __syncthreads();
  if (threadIdx.x == 0) {
    float s = 0.0f;
    for (int i = 0; i < CH; ++i) s += sdec[i];
    decc[b * NCH + n] = s * (1.0f / CH);
  }
}

// ===========================================================================
// gate = (silu(c) @ W_mod + b_mod)[:, 2D:3D]. grid = (D/256, B).
// ===========================================================================
__global__ void gate_kernel(const float* __restrict__ c,
                            const float* __restrict__ Wm,
                            const float* __restrict__ bm,
                            float* __restrict__ gate) {
  int b = (int)blockIdx.y;
  int j = (int)blockIdx.x * 256 + (int)threadIdx.x;
  __shared__ float sc[256];
  float acc = bm[2 * D + j];
  for (int i0 = 0; i0 < D; i0 += 256) {
    float cv = c[(size_t)b * D + i0 + threadIdx.x];
    sc[threadIdx.x] = cv * sigmoid_f(cv);
    __syncthreads();
    for (int ii = 0; ii < 256; ++ii)
      acc += sc[ii] * Wm[(size_t)(i0 + ii) * (3 * D) + 2 * D + j];
    __syncthreads();
  }
  gate[(size_t)b * D + j] = acc;
}

// ===========================================================================
// Scan P1: hq = gelu(q@W1) (bf16), a = gelu(k@W1) (bf16 rm + aat[e][c]),
// hk pre-act fp32. grid=(8,B); waves 0-3 q-path, 4-7 k-path; 2 tiles/wave.
// ===========================================================================
__global__ void scan_p1(const __bf16* __restrict__ qb,
                        const __bf16* __restrict__ kb,
                        const __bf16* __restrict__ w1t,
                        __bf16* __restrict__ hq,
                        __bf16* __restrict__ aa,
                        __bf16* __restrict__ aat,
                        float* __restrict__ hk, int step) {
  int b = (int)blockIdx.y;
  int wave = (int)(threadIdx.x >> 5);
  int tp = wave & 3, path = wave >> 2;
  int nb = (int)blockIdx.x * 128 + tp * 32;

  const __bf16* A = (path == 0 ? qb : kb) + ((size_t)b * S + (size_t)step * CH) * D;
  const __bf16* Wt0 = w1t + (size_t)b * D * D + (size_t)nb * D;
  const __bf16* Wt1 = Wt0 + (size_t)16 * D;

  f32x8 acc[2] = {zero8(), zero8()};
  for (int kk = 0; kk < D; kk += 32) {
    if (kk + 64 < D) __builtin_prefetch(Wt0 + kk + 64, 0, 3);
    bf16x16 af = load_a(A, D, kk);
    acc[0] = wmma_bf16(af, load_b(Wt0, D, kk), acc[0]);
    acc[1] = wmma_bf16(af, load_b(Wt1, D, kk), acc[1]);
  }
  int l = laneid(), mo = (l >> 4) * 8;
#pragma unroll
  for (int t = 0; t < 2; ++t) {
    int n0 = nb + t * 16 + (l & 15);
    f32x8 g;
#pragma unroll
    for (int r = 0; r < 8; ++r) g[r] = gelu_f(acc[t][r]);
    if (path == 0) {
      __bf16* HQ = hq + (size_t)b * CH * D;
#pragma unroll
      for (int r = 0; r < 8; ++r) HQ[(size_t)(mo + r) * D + n0] = (__bf16)g[r];
    } else {
      __bf16* AA = aa + (size_t)b * CH * D;
      float* HK = hk + (size_t)b * CH * D;
#pragma unroll
      for (int r = 0; r < 8; ++r) {
        HK[(size_t)(mo + r) * D + n0] = acc[t][r];
        AA[(size_t)(mo + r) * D + n0] = (__bf16)g[r];
      }
      *(bf16x8*)(aat + (size_t)b * D * CH + (size_t)n0 * CH + mo) = pack8(g);
    }
  }
}

// ===========================================================================
// Scan P2: retrieved = hq@W2 -> out = x + retrieved*gate     (path 0)
//          pred = a@W2 -> dpred = lr*(pred - v) (bf16 rm + dpt[f][c]) (path 1)
// grid=(8,B); 2 tiles/wave.
// ===========================================================================
__global__ void scan_p2(const __bf16* __restrict__ hq,
                        const __bf16* __restrict__ aa,
                        const __bf16* __restrict__ w2t,
                        const __bf16* __restrict__ vb,
                        const float* __restrict__ lr,
                        const float* __restrict__ x,
                        const float* __restrict__ gate,
                        float* __restrict__ out,
                        __bf16* __restrict__ dp,
                        __bf16* __restrict__ dpt, int step) {
  int b = (int)blockIdx.y;
  int wave = (int)(threadIdx.x >> 5);
  int tp = wave & 3, path = wave >> 2;
  int nb = (int)blockIdx.x * 128 + tp * 32;

  const __bf16* A = (path == 0 ? hq : aa) + (size_t)b * CH * D;
  const __bf16* Wt0 = w2t + (size_t)b * D * D + (size_t)nb * D;
  const __bf16* Wt1 = Wt0 + (size_t)16 * D;

  f32x8 acc[2] = {zero8(), zero8()};
  for (int kk = 0; kk < D; kk += 32) {
    if (kk + 64 < D) __builtin_prefetch(Wt0 + kk + 64, 0, 3);
    bf16x16 af = load_a(A, D, kk);
    acc[0] = wmma_bf16(af, load_b(Wt0, D, kk), acc[0]);
    acc[1] = wmma_bf16(af, load_b(Wt1, D, kk), acc[1]);
  }
  int l = laneid(), mo = (l >> 4) * 8;
#pragma unroll
  for (int t = 0; t < 2; ++t) {
    int n0 = nb + t * 16 + (l & 15);
    if (path == 0) {
      float gv = gate[(size_t)b * D + n0];
#pragma unroll
      for (int r = 0; r < 8; ++r) {
        size_t row = (size_t)b * S + (size_t)step * CH + mo + r;
        out[row * D + n0] = x[row * D + n0] + acc[t][r] * gv;
      }
    } else {
      const __bf16* V = vb + ((size_t)b * S + (size_t)step * CH) * D;
      __bf16* DP = dp + (size_t)b * CH * D;
      f32x8 dv;
#pragma unroll
      for (int r = 0; r < 8; ++r) {
        int m = mo + r;
        float ll = lr[(size_t)b * S + step * CH + m];
        dv[r] = ll * (acc[t][r] - (float)V[(size_t)m * D + n0]);
        DP[(size_t)m * D + n0] = (__bf16)dv[r];
      }
      *(bf16x8*)(dpt + (size_t)b * D * CH + (size_t)n0 * CH + mo) = pack8(dv);
    }
  }
}

// ===========================================================================
// Scan P3: da = dpred @ W2^T ; dhkt[e][c] = da * gelu'(hk). grid=(8,B),
// 4 waves, 2 tiles/wave. B contracts along row-contiguous f of w2rm[e][f].
// ===========================================================================
__global__ void scan_p3(const __bf16* __restrict__ dp,
                        const __bf16* __restrict__ w2rm,
                        const float* __restrict__ hk,
                        __bf16* __restrict__ dhkt, int step) {
  int b = (int)blockIdx.y;
  int wave = (int)(threadIdx.x >> 5);              // 0..3
  int nb = (int)blockIdx.x * 128 + wave * 32;      // e columns

  const __bf16* A = dp + (size_t)b * CH * D;
  const __bf16* Wr0 = w2rm + (size_t)b * D * D + (size_t)nb * D;
  const __bf16* Wr1 = Wr0 + (size_t)16 * D;

  f32x8 acc[2] = {zero8(), zero8()};
  for (int kk = 0; kk < D; kk += 32) {
    if (kk + 64 < D) __builtin_prefetch(Wr0 + kk + 64, 0, 3);
    bf16x16 af = load_a(A, D, kk);
    acc[0] = wmma_bf16(af, load_b(Wr0, D, kk), acc[0]);
    acc[1] = wmma_bf16(af, load_b(Wr1, D, kk), acc[1]);
  }
  int l = laneid(), mo = (l >> 4) * 8;
  const float* HK = hk + (size_t)b * CH * D;
#pragma unroll
  for (int t = 0; t < 2; ++t) {
    int n0 = nb + t * 16 + (l & 15);
    f32x8 dv;
#pragma unroll
    for (int r = 0; r < 8; ++r)
      dv[r] = acc[t][r] * dgelu_f(HK[(size_t)(mo + r) * D + n0]);
    *(bf16x8*)(dhkt + (size_t)b * D * CH + (size_t)n0 * CH + mo) = pack8(dv);
  }
}

// ===========================================================================
// Scan P4: rank-16 updates (K=16 padded in-register).
//   z=0: W1 = W1*keep - kt^T@dhk  -> fp32 master + w1t shadow
//   z=1: W2 = W2*keep - aat^T@dp  -> fp32 master + w2rm + w2t shadows
// grid = (8, B, 2); 8 waves; wave = two 16-col tiles (shared A) x 32 row tiles.
// ===========================================================================
__global__ void scan_p4(const __bf16* __restrict__ kt,
                        const __bf16* __restrict__ aat,
                        const __bf16* __restrict__ dhkt,
                        const __bf16* __restrict__ dpt,
                        const float* __restrict__ decc,
                        float* __restrict__ w1b, float* __restrict__ w2b,
                        __bf16* __restrict__ w1t,
                        __bf16* __restrict__ w2rm,
                        __bf16* __restrict__ w2t, int step) {
  int b = (int)blockIdx.y, z = (int)blockIdx.z;
  int wave = (int)(threadIdx.x >> 5);
  int nsel = wave & 3, mgrp = wave >> 2;           // 2 n-tiles, m split 2 ways
  int nb = (int)blockIdx.x * 128 + nsel * 32;
  float keep = 1.0f - decc[b * NCH + step];

  const __bf16* Bbase = (z == 0 ? dhkt : dpt) + (size_t)b * D * CH;
  bf16x16 bfrag0 = load_b_k16(Bbase + (size_t)nb * CH);
  bf16x16 bfrag1 = load_b_k16(Bbase + (size_t)(nb + 16) * CH);

  const __bf16* Asrc = (z == 0)
      ? kt + ((size_t)(b * NCH + step) * D) * CH
      : aat + (size_t)b * D * CH;
  float* Wm = (z == 0 ? w1b : w2b) + (size_t)b * D * D;
  size_t shoff = (size_t)b * D * D;

  int l = laneid(), mo = (l >> 4) * 8;
  for (int i = 0; i < 32; ++i) {
    int mt = i * 2 + mgrp;
    bf16x16 afrag = load_a_k16(Asrc + (size_t)mt * 16 * CH);
    f32x8 g0 = wmma_bf16(afrag, bfrag0, zero8());
    f32x8 g1 = wmma_bf16(afrag, bfrag1, zero8());
#pragma unroll
    for (int t = 0; t < 2; ++t) {
      const f32x8& g = t == 0 ? g0 : g1;
      int n0 = nb + t * 16 + (l & 15);
      f32x8 w;
#pragma unroll
      for (int r = 0; r < 8; ++r) {
        size_t idx = (size_t)(mt * 16 + mo + r) * D + n0;
        w[r] = Wm[idx] * keep - g[r];
        Wm[idx] = w[r];
      }
      if (z == 0) {
        *(bf16x8*)(w1t + shoff + (size_t)n0 * D + mt * 16 + mo) = pack8(w);
      } else {
#pragma unroll
        for (int r = 0; r < 8; ++r)
          w2rm[shoff + (size_t)(mt * 16 + mo + r) * D + n0] = (__bf16)w[r];
        *(bf16x8*)(w2t + shoff + (size_t)n0 * D + mt * 16 + mo) = pack8(w);
      }
    }
  }
}

// ===========================================================================
// Host launcher
// ===========================================================================
extern "C" void kernel_launch(void* const* d_in, const int* in_sizes, int n_in,
                              void* d_out, int out_size, void* d_ws, size_t ws_size,
                              hipStream_t stream) {
  (void)in_sizes; (void)n_in; (void)out_size; (void)ws_size;

  const float* x    = (const float*)d_in[0];
  const float* c    = (const float*)d_in[1];
  const float* Wq   = (const float*)d_in[2];
  const float* Wk   = (const float*)d_in[3];
  const float* Wv   = (const float*)d_in[4];
  const float* wlr  = (const float*)d_in[5];
  const float* wdec = (const float*)d_in[6];
  const float* W1   = (const float*)d_in[7];
  const float* W2   = (const float*)d_in[8];
  const float* Wm   = (const float*)d_in[9];
  const float* bm   = (const float*)d_in[10];
  float* out = (float*)d_out;

  char* p = (char*)d_ws;
  auto carve = [&](size_t bytes) -> char* {
    char* r = p;
    p += (bytes + 255) & ~(size_t)255;
    return r;
  };
  const size_t BSD = (size_t)BB * S * D;
  const size_t DD  = (size_t)D * D;
  __bf16* xb   = (__bf16*)carve(BSD * 2);
  __bf16* Wt   = (__bf16*)carve(3 * DD * 2);
  __bf16* qb   = (__bf16*)carve(BSD * 2);
  __bf16* kb   = (__bf16*)carve(BSD * 2);
  __bf16* vb   = (__bf16*)carve(BSD * 2);
  __bf16* kt   = (__bf16*)carve(BSD * 2);
  float*  w1b  = (float*)carve(BB * DD * 4);
  float*  w2b  = (float*)carve(BB * DD * 4);
  __bf16* w1t  = (__bf16*)carve(BB * DD * 2);
  __bf16* w2rm = (__bf16*)carve(BB * DD * 2);
  __bf16* w2t  = (__bf16*)carve(BB * DD * 2);
  float*  lrp  = (float*)carve((size_t)BB * S * 4);
  float*  decc = (float*)carve((size_t)BB * NCH * 4);
  float*  gate = (float*)carve((size_t)BB * D * 4);
  __bf16* hq   = (__bf16*)carve((size_t)BB * CH * D * 2);
  __bf16* aa   = (__bf16*)carve((size_t)BB * CH * D * 2);
  __bf16* aat  = (__bf16*)carve((size_t)BB * CH * D * 2);
  float*  hk   = (float*)carve((size_t)BB * CH * D * 4);
  __bf16* dp   = (__bf16*)carve((size_t)BB * CH * D * 2);
  __bf16* dpt  = (__bf16*)carve((size_t)BB * CH * D * 2);
  __bf16* dhkt = (__bf16*)carve((size_t)BB * CH * D * 2);

  cvt_x_kernel<<<dim3(BSD / 2048), 256, 0, stream>>>(x, xb);
  wt_kernel<<<dim3(DD / 256, 3), 256, 0, stream>>>(Wq, Wk, Wv, Wt);
  init_w_kernel<<<dim3(DD / 256, BB, 2), 256, 0, stream>>>(W1, W2, w1b, w2b,
                                                           w1t, w2rm, w2t);
  proj_kernel<<<dim3((BB * S) / 128, D / 64, 3), 256, 0, stream>>>(
      xb, Wt, qb, kb, vb, kt);
  lrdecay_kernel<<<dim3(BB * NCH), 256, 0, stream>>>(x, wlr, wdec, lrp, decc);
  gate_kernel<<<dim3(D / 256, BB), 256, 0, stream>>>(c, Wm, bm, gate);

  for (int n = 0; n < NCH; ++n) {
    scan_p1<<<dim3(8, BB), 256, 0, stream>>>(qb, kb, w1t, hq, aa, aat, hk, n);
    scan_p2<<<dim3(8, BB), 256, 0, stream>>>(hq, aa, w2t, vb, lrp, x, gate,
                                             out, dp, dpt, n);
    scan_p3<<<dim3(8, BB), 128, 0, stream>>>(dp, w2rm, hk, dhkt, n);
    scan_p4<<<dim3(8, BB, 2), 256, 0, stream>>>(kt, aat, dhkt, dpt, decc,
                                                w1b, w2b, w1t, w2rm, w2t, n);
  }
}
